// TopKPool_56573309223699
// MI455X (gfx1250) — matrix operand order repaired and checked by reading, not compiled, not run on hardware
//
#include <hip/hip_runtime.h>
#include <hip/hip_bf16.h>
#include <math.h>

// ---------------- problem constants (match reference) ----------------
constexpr int NUM_GRAPHS      = 64;
constexpr int NPG             = 2048;                 // nodes per graph
constexpr int NUM_NODES       = NUM_GRAPHS * NPG;     // 131072
constexpr int D               = 256;
constexpr long NUM_EDGES      = (long)NUM_NODES * 32; // 4194304
constexpr int TOPK            = NPG / 2;              // 1024
constexpr int POOLED          = NUM_GRAPHS * TOPK;    // 65536

// output layout (flat f32, concatenated in reference return order)
constexpr long X_OFF = 0;                              // x_pooled      [65536,256]
constexpr long E_OFF = X_OFF + (long)POOLED * D;       // edge_index_pooled [2,E]
constexpr long B_OFF = E_OFF + 2 * NUM_EDGES;          // batch_pooled  [65536]
constexpr long P_OFF = B_OFF + POOLED;                 // perm          [65536]
constexpr long S_OFF = P_OFF + POOLED;                 // scores[perm]  [65536]
constexpr long M_OFF = S_OFF + POOLED;                 // edge_mask     [E]

typedef __attribute__((ext_vector_type(16))) _Float16 v16h;
typedef __attribute__((ext_vector_type(8)))  float    v8f;

// ---------------------------------------------------------------------
// K1: scores = tanh(x @ W + b) via v_wmma_f32_16x16x32_f16.
// One wave computes 16 rows; 8 WMMAs sweep K=256. B is column-constant
// (B[k][n] = W[k] for all n), so only the K mapping matters for B.
// ---------------------------------------------------------------------
__global__ __launch_bounds__(256) void scores_wmma_kernel(
    const float* __restrict__ x, const float* __restrict__ W,
    const float* __restrict__ b, float* __restrict__ scores)
{
    const int lane    = threadIdx.x & 31;
    const int wave    = threadIdx.x >> 5;
    const int rowBase = (blockIdx.x * 8 + wave) * 16;   // 16 rows per wave
    const int half    = lane >> 4;                      // 0: lanes 0-15, 1: lanes 16-31
    const int m       = lane & 15;
    const long rowA   = (long)rowBase + m;

    v8f acc = {};
    #pragma unroll
    for (int kc = 0; kc < D; kc += 32) {
        v16h a, bb;
        #pragma unroll
        for (int p = 0; p < 8; ++p) {
            // 16-bit A 16x32 layout: VGPR p holds K = 2p(+8 if p>=4) + 8*half, +1
            const int colA = kc + 2 * p + ((p >= 4) ? 8 : 0) + 8 * half;
            const float2 va = *(const float2*)(x + rowA * D + colA);
            a[2 * p]     = (_Float16)va.x;
            a[2 * p + 1] = (_Float16)va.y;
            // B 32x16: VGPR p holds K = 2p(+8 if p>=4) + 8*half, +1 (value independent of N)
            const int colB = kc + 2 * p + ((p >= 4) ? 8 : 0) + 8 * half;
            const float2 vb = *(const float2*)(W + colB);
            bb[2 * p]     = (_Float16)vb.x;
            bb[2 * p + 1] = (_Float16)vb.y;
        }
        acc = __builtin_amdgcn_wmma_f32_16x16x32_f16(
            /*neg_a=*/false, a, /*neg_b=*/false, bb,
            /*c_mod=*/(short)0, acc, /*reuse_a=*/false, /*reuse_b=*/false);
    }

    // D layout: VGPR r, lanes 0-15 -> (M=r, N=lane); lanes 16-31 -> (M=r+8).
    // Column N=0 lives in lane 0 (M=0..7) and lane 16 (M=8..15).
    const float bias = b[0];
    if ((lane & 15) == 0) {
        const int mbase = rowBase + half * 8;
        #pragma unroll
        for (int r = 0; r < 8; ++r)
            scores[mbase + r] = tanhf(acc[r] + bias);
    }
}

// ---------------------------------------------------------------------
// K2: per-graph full bitonic sort (descending, index tie-break) of 2048
// (score, idx) pairs in LDS. First TOPK ranks are the sorted top-k.
// Every node's survive-mask is written exactly once (no zero-init needed).
// ---------------------------------------------------------------------
__global__ __launch_bounds__(1024) void topk_sort_kernel(
    const float* __restrict__ scores,
    int* __restrict__ perm_ws, unsigned char* __restrict__ node_mask,
    float* __restrict__ out_perm, float* __restrict__ out_scores,
    float* __restrict__ out_batch)
{
    __shared__ float s[NPG];
    __shared__ int   id[NPG];

    const int g    = blockIdx.x;
    const int tid  = threadIdx.x;
    const int base = g * NPG;

    for (int i = tid; i < NPG; i += 1024) { s[i] = scores[base + i]; id[i] = i; }
    __syncthreads();

    for (int ksize = 2; ksize <= NPG; ksize <<= 1) {
        for (int j = ksize >> 1; j > 0; j >>= 1) {
            for (int i = tid; i < NPG; i += 1024) {
                const int p = i ^ j;
                if (p > i) {
                    const bool desc = ((i & ksize) == 0);
                    const float si = s[i], sp = s[p];
                    const int   ii = id[i], ip = id[p];
                    // strict total order: score desc, then index asc
                    const bool pBeforeI = (sp > si) || (sp == si && ip < ii);
                    if (desc ? pBeforeI : !pBeforeI) {
                        s[i] = sp; s[p] = si; id[i] = ip; id[p] = ii;
                    }
                }
            }
            __syncthreads();
        }
    }

    for (int r = tid; r < NPG; r += 1024) {
        const int gi = base + id[r];
        node_mask[gi] = (r < TOPK) ? 1 : 0;
        if (r < TOPK) {
            const int o   = g * TOPK + r;
            perm_ws[o]    = gi;
            out_perm[o]   = (float)gi;
            out_scores[o] = s[r];
            out_batch[o]  = (float)g;
        }
    }
}

// ---------------------------------------------------------------------
// K3: x_pooled = x[perm]  (coalesced float4 row gather, 67 MB each way)
// ---------------------------------------------------------------------
__global__ __launch_bounds__(256) void gather_rows_kernel(
    const float* __restrict__ x, const int* __restrict__ perm,
    float* __restrict__ outx)
{
    const long t   = (long)blockIdx.x * blockDim.x + threadIdx.x; // POOLED*64 threads
    const int  row = (int)(t >> 6);
    const int  off = (int)(t & 63) << 2;
    const long src = (long)perm[row];
    *(float4*)(outx + (long)row * D + off) = *(const float4*)(x + src * D + off);
}

// ---------------------------------------------------------------------
// K4: edge mask + masked edge_index (dropped edges -> -1), mask as 0/1 f32
// ---------------------------------------------------------------------
__global__ __launch_bounds__(256) void edge_mask_kernel(
    const int* __restrict__ ei, const unsigned char* __restrict__ node_mask,
    float* __restrict__ outE, float* __restrict__ outM)
{
    const long e = (long)blockIdx.x * blockDim.x + threadIdx.x;
    if (e >= NUM_EDGES) return;
    const int r = ei[e];
    const int c = ei[NUM_EDGES + e];
    const bool keep = node_mask[r] && node_mask[c];
    outE[e]             = keep ? (float)r : -1.0f;
    outE[NUM_EDGES + e] = keep ? (float)c : -1.0f;
    outM[e]             = keep ? 1.0f : 0.0f;
}

// ---------------------------------------------------------------------
extern "C" void kernel_launch(void* const* d_in, const int* in_sizes, int n_in,
                              void* d_out, int out_size, void* d_ws, size_t ws_size,
                              hipStream_t stream)
{
    const float* x  = (const float*)d_in[0];   // [131072, 256]
    const int*   ei = (const int*)  d_in[1];   // [2, 4194304]
    // d_in[2] = batch (unused: graphs are contiguous & equal-sized)
    const float* W  = (const float*)d_in[3];   // [256, 1]
    const float* b  = (const float*)d_in[4];   // [1]

    float* out = (float*)d_out;

    // workspace: perm (int) then node mask (bytes)
    int*           perm_ws   = (int*)d_ws;                            // 65536 * 4 B
    unsigned char* node_mask = (unsigned char*)d_ws + (size_t)POOLED * 4; // 131072 B

    // scores staged in the x_pooled output region (consumed by K2 before K3
    // overwrites it) — keeps workspace footprint at 384 KB.
    float* scores_tmp = out + X_OFF;

    // K1: WMMA mat-vec scores (1024 blocks x 8 waves x 16 rows = 131072 rows)
    scores_wmma_kernel<<<NUM_NODES / 128, 256, 0, stream>>>(x, W, b, scores_tmp);

    // K2: per-graph bitonic top-k (one 1024-thread block per graph)
    topk_sort_kernel<<<NUM_GRAPHS, 1024, 0, stream>>>(
        scores_tmp, perm_ws, node_mask,
        out + P_OFF, out + S_OFF, out + B_OFF);

    // K3: row gather
    gather_rows_kernel<<<(POOLED * 64) / 256, 256, 0, stream>>>(x, perm_ws, out + X_OFF);

    // K4: edge masking
    edge_mask_kernel<<<(int)((NUM_EDGES + 255) / 256), 256, 0, stream>>>(
        ei, node_mask, out + E_OFF, out + M_OFF);
}
